// HGTLayer_89000312307818
// MI455X (gfx1250) — matrix operand order repaired and checked by reading, not compiled, not run on hardware
//
#include <hip/hip_runtime.h>
#include <hip/hip_bf16.h>
#include <cstdint>
#include <cstddef>

typedef __bf16 v16bf __attribute__((ext_vector_type(16)));
typedef __bf16 v8bf  __attribute__((ext_vector_type(8)));
typedef float  v8f   __attribute__((ext_vector_type(8)));
typedef float  v4f   __attribute__((ext_vector_type(4)));

#define NNODES 100000
#define NEDGES 500000
#define DIM    128
#define NREL   3
#define NEG_SLOPE 0.2f
#define OT_PITCH 20   // padded row pitch (floats) for epilogue LDS tile: 16B aligned, no bank conflicts

// ---------------- helpers ----------------

static __device__ __forceinline__ unsigned short f32_to_bf16_rne(float f) {
    unsigned u = __float_as_uint(f);
    u += 0x7FFFu + ((u >> 16) & 1u);
    return (unsigned short)(u >> 16);
}

// order-preserving float<->uint map so segment-max can use integer atomicMax
static __device__ __forceinline__ unsigned enc_f32(float v) {
    unsigned u = __float_as_uint(v);
    return (u & 0x80000000u) ? ~u : (u | 0x80000000u);
}
static __device__ __forceinline__ float dec_f32(unsigned u) {
    return __uint_as_float((u & 0x80000000u) ? (u ^ 0x80000000u) : ~u);
}

// ---------------- kernels ----------------

__global__ void convert_bf16_kernel(const float* __restrict__ in,
                                    unsigned short* __restrict__ out, int n) {
    int i = blockIdx.x * blockDim.x + threadIdx.x;
    if (i < n) out[i] = f32_to_bf16_rne(in[i]);
}

// out[n,d] = mean_r bias[r][d]  (accumulation target for the 3 relations)
__global__ void init_out_kernel(const float* __restrict__ bias,
                                float* __restrict__ out, int total) {
    int i = blockIdx.x * blockDim.x + threadIdx.x;
    if (i >= total) return;
    int d = i & (DIM - 1);
    out[i] = (bias[d] + bias[DIM + d] + bias[2 * DIM + d]) * (1.0f / 3.0f);
}

// feat = x @ W^T via v_wmma_f32_16x16x32_bf16.
//  - W[r] (32 KB bf16) is staged once per workgroup into LDS; B fragments come
//    from ds_load_b128 instead of 8x-redundant global traffic.
//  - One wave owns a 16-row tile of x; the four K=32 A-fragments live in VGPRs
//    and are reused across all 8 column tiles (32 WMMAs / wave).
//  - Epilogue transposes each 16x16 f32 tile through a per-wave LDS buffer so
//    feat is written with coalesced global_store_b128.
__global__ void __launch_bounds__(256)
gemm_wmma_kernel(const unsigned short* __restrict__ xh,   // [N,128] bf16
                 const unsigned short* __restrict__ wh,   // [128,128] bf16 (row d, col k)
                 float* __restrict__ feat, int nTiles) {
    __shared__ unsigned short wlds[DIM * DIM];            // 32 KB weight stage
    __shared__ float otile[8][16 * OT_PITCH];             // 8 waves x 1.25 KB epilogue stage

    // ---- cooperative W stage (all 256 threads, before any early-exit) ----
    {
        const uint4* wsrc = (const uint4*)wh;
        uint4* wdst = (uint4*)wlds;
#pragma unroll
        for (int t = 0; t < (DIM * DIM) / (8 * 256); ++t)   // 2048 uint4 / 256 thr = 8
            wdst[threadIdx.x + t * 256] = wsrc[threadIdx.x + t * 256];
    }
    __syncthreads();

    int lane = threadIdx.x & 31;
    int wave = threadIdx.x >> 5;
    int tile = blockIdx.x * 8 + wave;
    if (tile >= nTiles) return;                 // wave-uniform exit (EXEC all-1 for WMMA)

    int row0  = tile * 16;
    int m     = lane & 15;
    int khalf = (lane >> 4) * 8;                // ISA 16-bit A-matrix swizzle

    const unsigned short* xrow = xh + (size_t)(row0 + m) * DIM;
    v16bf a[4];
#pragma unroll
    for (int ks = 0; ks < 4; ++ks) {
        v8bf lo = *(const v8bf*)(xrow + ks * 32 + khalf);        // K = ks*32+khalf+0..7
        v8bf hi = *(const v8bf*)(xrow + ks * 32 + 16 + khalf);   // K = ks*32+16+khalf+0..7
        a[ks] = __builtin_shufflevector(lo, hi,
                    0, 1, 2, 3, 4, 5, 6, 7, 8, 9, 10, 11, 12, 13, 14, 15);
    }

    int nl    = lane & 15;
    int kb    = (lane >> 4) * 8;                // B: same K swizzle as A (khalf, 16+khalf)
    int mhalf = (lane >> 4) * 8;                // C/D: VGPR i = row i (+8 for hi lanes)
    float* tp = &otile[wave][0];

    int srow = lane >> 1;                       // epilogue: lane pair per output row
    int scol = (lane & 1) * 8;

#pragma unroll
    for (int nt = 0; nt < 8; ++nt) {
        const unsigned short* wrow = wlds + (nt * 16 + nl) * DIM + kb;
        v8f acc = {};
#pragma unroll
        for (int ks = 0; ks < 4; ++ks) {
            v8bf blo = *(const v8bf*)(wrow + ks * 32);
            v8bf bhi = *(const v8bf*)(wrow + ks * 32 + 16);
            v16bf b = __builtin_shufflevector(blo, bhi,
                    0, 1, 2, 3, 4, 5, 6, 7, 8, 9, 10, 11, 12, 13, 14, 15);
            acc = __builtin_amdgcn_wmma_f32_16x16x32_bf16(
                      false, a[ks], false, b, (short)0, acc, false, false);
        }
        // transpose 16x16 tile through LDS (wave-synchronous, DS is in-order)
#pragma unroll
        for (int i = 0; i < 8; ++i)
            tp[(mhalf + i) * OT_PITCH + nl] = acc[i];
        v4f c0 = *(const v4f*)(tp + srow * OT_PITCH + scol);
        v4f c1 = *(const v4f*)(tp + srow * OT_PITCH + scol + 4);
        float* gp = feat + (size_t)(row0 + srow) * DIM + nt * 16 + scol;
        *(v4f*)gp       = c0;                   // global_store_b128
        *(v4f*)(gp + 4) = c1;
    }
}

// el[n] = feat[n]·al, er[n] = feat[n]·ar; also re-init per-relation max/sum
__global__ void node_attn_kernel(const float* __restrict__ feat,
                                 const float* __restrict__ al,
                                 const float* __restrict__ ar,
                                 float* __restrict__ el, float* __restrict__ er,
                                 unsigned* __restrict__ menc, float* __restrict__ s,
                                 int n) {
    int i = blockIdx.x * blockDim.x + threadIdx.x;
    if (i >= n) return;
    const float* fr = feat + (size_t)i * DIM;
    float a = 0.f, b = 0.f;
#pragma unroll 4
    for (int d = 0; d < DIM; d += 4) {
        v4f f = *(const v4f*)(fr + d);
        v4f l = *(const v4f*)(al + d);
        v4f r = *(const v4f*)(ar + d);
        a += f[0] * l[0] + f[1] * l[1] + f[2] * l[2] + f[3] * l[3];
        b += f[0] * r[0] + f[1] * r[1] + f[2] * r[2] + f[3] * r[3];
    }
    el[i] = a;
    er[i] = b;
    menc[i] = 0u;      // encoded -inf
    s[i] = 0.f;
}

// e = leaky_relu(el[src]+er[dst]); segment-max over dst via integer atomicMax
__global__ void edge_score_kernel(const int* __restrict__ src, const int* __restrict__ dst,
                                  const float* __restrict__ el, const float* __restrict__ er,
                                  float* __restrict__ ebuf, unsigned* __restrict__ menc,
                                  int ne) {
    int e = blockIdx.x * blockDim.x + threadIdx.x;
    if (e >= ne) return;
    int d = dst[e];
    float v = el[src[e]] + er[d];
    v = v > 0.f ? v : NEG_SLOPE * v;
    ebuf[e] = v;
    atomicMax(menc + d, enc_f32(v));
}

// ex = exp(e - m[dst]); segment-sum over dst
__global__ void edge_exp_kernel(const int* __restrict__ dst,
                                const float* __restrict__ ebuf,
                                const unsigned* __restrict__ menc,
                                float* __restrict__ exbuf, float* __restrict__ s,
                                int ne) {
    int e = blockIdx.x * blockDim.x + threadIdx.x;
    if (e >= ne) return;
    int d = dst[e];
    float ex = __expf(ebuf[e] - dec_f32(menc[d]));
    exbuf[e] = ex;
    atomicAdd(s + d, ex);
}

// out[dst] += (alpha/3) * feat[src] ; one wave per edge, lane covers 4 channels
__global__ void __launch_bounds__(256)
edge_scatter_kernel(const int* __restrict__ src, const int* __restrict__ dst,
                    const float* __restrict__ exbuf, const float* __restrict__ s,
                    const float* __restrict__ feat, float* __restrict__ out,
                    int ne) {
    int lane = threadIdx.x & 31;
    int e = (blockIdx.x * blockDim.x + threadIdx.x) >> 5;
    if (e >= ne) return;
    int si = src[e], di = dst[e];
    float coef = exbuf[e] / s[di] * (1.0f / 3.0f);
    v4f f = *(const v4f*)(feat + (size_t)si * DIM + lane * 4);
    float* orow = out + (size_t)di * DIM + lane * 4;
    atomicAdd(orow + 0, coef * f[0]);
    atomicAdd(orow + 1, coef * f[1]);
    atomicAdd(orow + 2, coef * f[2]);
    atomicAdd(orow + 3, coef * f[3]);
}

// ---------------- launcher ----------------

extern "C" void kernel_launch(void* const* d_in, const int* in_sizes, int n_in,
                              void* d_out, int out_size, void* d_ws, size_t ws_size,
                              hipStream_t stream) {
    (void)in_sizes; (void)n_in; (void)out_size; (void)ws_size;

    const float* x    = (const float*)d_in[0];
    const int*   src  = (const int*)  d_in[1];
    const int*   dst  = (const int*)  d_in[2];
    const float* W    = (const float*)d_in[3];
    const float* al   = (const float*)d_in[4];
    const float* ar   = (const float*)d_in[5];
    const float* bias = (const float*)d_in[6];
    float*       out  = (float*)d_out;

    // scratch carving (256B aligned slots)
    char* ws = (char*)d_ws;
    size_t off = 0;
    auto carve = [&](size_t bytes) -> char* {
        char* p = ws + off;
        off = (off + bytes + 255) & ~(size_t)255;
        return p;
    };
    unsigned short* xh   = (unsigned short*)carve((size_t)NNODES * DIM * 2);
    unsigned short* wh   = (unsigned short*)carve((size_t)NREL * DIM * DIM * 2);
    float*          feat = (float*)carve((size_t)NNODES * DIM * 4);
    float*          el   = (float*)carve((size_t)NNODES * 4);
    float*          er   = (float*)carve((size_t)NNODES * 4);
    unsigned*       menc = (unsigned*)carve((size_t)NNODES * 4);
    float*          ssum = (float*)carve((size_t)NNODES * 4);
    float*          ebuf = (float*)carve((size_t)NEDGES * 4);
    float*          exbf = (float*)carve((size_t)NEDGES * 4);

    const int nx = NNODES * DIM;            // 12.8M
    const int nw = NREL * DIM * DIM;        // 49152
    const int nTiles = NNODES / 16;         // 6250 (exact)

    convert_bf16_kernel<<<(nx + 255) / 256, 256, 0, stream>>>(x, xh, nx);
    convert_bf16_kernel<<<(nw + 255) / 256, 256, 0, stream>>>(W, wh, nw);
    init_out_kernel<<<(nx + 255) / 256, 256, 0, stream>>>(bias, out, nx);

    for (int r = 0; r < NREL; ++r) {
        const unsigned short* whr = wh + (size_t)r * DIM * DIM;
        gemm_wmma_kernel<<<(nTiles + 7) / 8, 256, 0, stream>>>(xh, whr, feat, nTiles);
        node_attn_kernel<<<(NNODES + 255) / 256, 256, 0, stream>>>(
            feat, al + r * DIM, ar + r * DIM, el, er, menc, ssum, NNODES);
        edge_score_kernel<<<(NEDGES + 255) / 256, 256, 0, stream>>>(
            src + (size_t)r * NEDGES, dst + (size_t)r * NEDGES, el, er, ebuf, menc, NEDGES);
        edge_exp_kernel<<<(NEDGES + 255) / 256, 256, 0, stream>>>(
            dst + (size_t)r * NEDGES, ebuf, menc, exbf, ssum, NEDGES);
        edge_scatter_kernel<<<(NEDGES * 32 + 255) / 256, 256, 0, stream>>>(
            src + (size_t)r * NEDGES, dst + (size_t)r * NEDGES, exbf, ssum, feat, out, NEDGES);
    }
}
